// ChannelMerger_46256797778500
// MI455X (gfx1250) — matrix operand order repaired and checked by reading, not compiled, not run on hardware
//
#include <hip/hip_runtime.h>

// Problem dims (fixed by reference)
#define B_ 32
#define C_ 512
#define T_ 2048
#define O_ 256
#define D_ 2048          // EMBED_DIM
#define NFREQ 32
#define MARGIN_F 0.2f
#define INVALID_F -0.1f

typedef __attribute__((ext_vector_type(16))) __bf16 v16bf;
typedef __attribute__((ext_vector_type(8)))  __bf16 v8bf;
typedef __attribute__((ext_vector_type(4)))  __bf16 v4bf;
typedef __attribute__((ext_vector_type(8)))  float  v8f;
typedef __attribute__((ext_vector_type(4)))  float  v4f;

union V16U { v16bf v; v8bf h[2]; };

// ---------------------------------------------------------------------------
// Kernel 1: convert x (f32) -> xb (bf16), 4 elements per thread
// ---------------------------------------------------------------------------
__global__ void __launch_bounds__(256) k_convert_x(const float* __restrict__ x,
                                                   __bf16* __restrict__ xb, int n4) {
  int i = blockIdx.x * blockDim.x + threadIdx.x;
  if (i >= n4) return;
  v4f f = *(const v4f*)(x + (size_t)i * 4);
  v4bf o;
  o[0] = (__bf16)f.x; o[1] = (__bf16)f.y; o[2] = (__bf16)f.z; o[3] = (__bf16)f.w;
  *(v4bf*)(xb + (size_t)i * 4) = o;
}

// ---------------------------------------------------------------------------
// Kernel 2: heads [O,D] f32 -> ht [D,O] bf16 (transpose; coalesced writes)
// ---------------------------------------------------------------------------
__global__ void __launch_bounds__(256) k_heads_t(const float* __restrict__ heads,
                                                 __bf16* __restrict__ ht, int n) {
  int i = blockIdx.x * blockDim.x + threadIdx.x;
  if (i >= n) return;
  int d = i / O_, o = i % O_;
  ht[i] = (__bf16)heads[(size_t)o * D_ + d];
}

// ---------------------------------------------------------------------------
// Kernel 3: fourier embedding -> embb [B*C, D] bf16; offs[b*C+c] = 0 / -inf
// One block per (b,c) row; each thread produces 4 (cos,sin) pairs.
// ---------------------------------------------------------------------------
__global__ void __launch_bounds__(256) k_embed(const float* __restrict__ pos,
                                               __bf16* __restrict__ embb,
                                               float* __restrict__ offs) {
  int bc = blockIdx.x;
  float px = pos[bc * 2 + 0];
  float py = pos[bc * 2 + 1];
  if (threadIdx.x == 0)
    offs[bc] = (px == INVALID_F && py == INVALID_F) ? -__builtin_inff() : 0.0f;

  const float scale = 6.283185307179586f / 1.4f;   // 2*pi / (1 + 2*margin)
  float ax = (px + MARGIN_F) * scale;
  float ay = (py + MARGIN_F) * scale;
  __bf16* row = embb + (size_t)bc * D_;
#pragma unroll
  for (int u = 0; u < 4; ++u) {
    int k = u * 256 + threadIdx.x;                  // 0..1023
    int fi = k >> 5, fj = k & 31;                   // freq indices
    float loc = ax * (float)fi + ay * (float)fj;
    float s, c;
    __sincosf(loc, &s, &c);
    row[k]        = (__bf16)c;                      // cos half
    row[1024 + k] = (__bf16)s;                      // sin half
  }
}

// ---------------------------------------------------------------------------
// Kernel 4: scores GEMM (S^T = E * H^T) + softmax over C, fused.
// Grid: B * (O/16) blocks of 256 threads (8 waves).
// Block covers ALL 512 c rows (8 waves x 4 m-tiles) for one 16-wide o slice.
// ---------------------------------------------------------------------------
__global__ void __launch_bounds__(256) k_scores(const __bf16* __restrict__ embb,
                                                const __bf16* __restrict__ ht,
                                                const float* __restrict__ offs,
                                                __bf16* __restrict__ wt) {
  const int b    = blockIdx.x >> 4;          // O_/16 == 16 o-tiles
  const int o0   = (blockIdx.x & 15) * 16;
  const int wave = threadIdx.x >> 5;
  const int lane = threadIdx.x & 31;
  const int ln16 = lane & 15;
  const int hi   = lane >> 4;                // 0 or 1 (lane half)

  const __bf16* E = embb + (size_t)b * C_ * D_;

  v8f acc[4];
#pragma unroll
  for (int mt = 0; mt < 4; ++mt) acc[mt] = (v8f){};

  for (int k0 = 0; k0 < D_; k0 += 32) {
    // B tile (32xd x 16xo): lane = o column ln16, K 0-7/16-23 vs 8-15/24-31 by lane half
    V16U Bm;
    const __bf16* bp = ht + (size_t)(k0 + hi * 8) * O_ + o0 + ln16;
#pragma unroll
    for (int s = 0; s < 8; ++s) { Bm.v[s] = bp[(size_t)s * O_]; Bm.v[8 + s] = bp[(size_t)(16 + s) * O_]; }
#pragma unroll
    for (int mt = 0; mt < 4; ++mt) {
      // A tile (16xc x 32xd): lane half selects K 0-7/16-23 vs 8-15/24-31
      int crow = wave * 64 + mt * 16 + ln16;
      const __bf16* ap = E + (size_t)crow * D_ + k0 + hi * 8;
      V16U Am;
      Am.h[0] = *(const v8bf*)(ap);
      Am.h[1] = *(const v8bf*)(ap + 16);
      acc[mt] = __builtin_amdgcn_wmma_f32_16x16x32_bf16(
          false, Am.v, false, Bm.v, (short)0, acc[mt], false, false);
    }
  }

  // apply -inf offset for invalid channels (c runs along accumulator rows)
  const float* offb = offs + b * C_;
#pragma unroll
  for (int mt = 0; mt < 4; ++mt) {
    int cbase = wave * 64 + mt * 16 + hi * 8;
    v4f oa = *(const v4f*)(offb + cbase);
    v4f ob = *(const v4f*)(offb + cbase + 4);
#pragma unroll
    for (int r = 0; r < 4; ++r) { acc[mt][r] += oa[r]; acc[mt][r + 4] += ob[r]; }
  }

  // softmax over C (rows) per o column (column = ln16, same for both halves)
  __shared__ float redmax[8][16];
  __shared__ float redsum[8][16];

  float m = -__builtin_inff();
#pragma unroll
  for (int mt = 0; mt < 4; ++mt)
#pragma unroll
    for (int r = 0; r < 8; ++r) m = fmaxf(m, acc[mt][r]);
  m = fmaxf(m, __shfl_xor(m, 16, 32));
  if (hi == 0) redmax[wave][ln16] = m;
  __syncthreads();
  float cm = -__builtin_inff();
#pragma unroll
  for (int w = 0; w < 8; ++w) cm = fmaxf(cm, redmax[w][ln16]);

  float s = 0.0f;
#pragma unroll
  for (int mt = 0; mt < 4; ++mt)
#pragma unroll
    for (int r = 0; r < 8; ++r) {
      float e = __expf(acc[mt][r] - cm);   // exp(-inf - finite) = 0 for invalid c
      acc[mt][r] = e;
      s += e;
    }
  s += __shfl_xor(s, 16, 32);
  if (hi == 0) redsum[wave][ln16] = s;
  __syncthreads();
  float cs = 0.0f;
#pragma unroll
  for (int w = 0; w < 8; ++w) cs += redsum[w][ln16];
  float inv = 1.0f / cs;

  // store weights wt[b, o0+ln16, c] — each lane owns 8 contiguous c per tile
  __bf16* wrow = wt + ((size_t)b * O_ + (o0 + ln16)) * C_;
#pragma unroll
  for (int mt = 0; mt < 4; ++mt) {
    int cbase = wave * 64 + mt * 16 + hi * 8;
    v8bf ob;
#pragma unroll
    for (int r = 0; r < 8; ++r) ob[r] = (__bf16)(acc[mt][r] * inv);
    *(v8bf*)(wrow + cbase) = ob;
  }
}

// ---------------------------------------------------------------------------
// Kernel 5: out[b,o,t] = sum_c wt[b,o,c] * xb[b,c,t]   (bf16 WMMA, f32 out)
// Grid: (T/128, O/64, B). Each wave owns a 64o x 16t strip: 4 o-tiles share
// one X B-tile per k-step (4 WMMA per B load; 4x less X re-read vs 16o blocks)
// ---------------------------------------------------------------------------
__global__ void __launch_bounds__(256) k_out(const __bf16* __restrict__ wt,
                                             const __bf16* __restrict__ xb,
                                             float* __restrict__ out) {
  const int b    = blockIdx.z;
  const int o0   = blockIdx.y * 64;
  const int wave = threadIdx.x >> 5;
  const int lane = threadIdx.x & 31;
  const int ln16 = lane & 15;
  const int hi   = lane >> 4;
  const int t0   = blockIdx.x * 128 + wave * 16;

  const __bf16* W = wt + (size_t)b * O_ * C_;
  const __bf16* X = xb + (size_t)b * C_ * T_;

  v8f acc[4];
#pragma unroll
  for (int ot = 0; ot < 4; ++ot) acc[ot] = (v8f){};

  for (int c0 = 0; c0 < C_; c0 += 32) {
    // B tile (32xc x 16xt): lane = t column ln16, K 0-7/16-23 vs 8-15/24-31 by lane half
    V16U Bm;
    const __bf16* bp = X + (size_t)(c0 + hi * 8) * T_ + t0 + ln16;
#pragma unroll
    for (int s = 0; s < 8; ++s) { Bm.v[s] = bp[(size_t)s * T_]; Bm.v[8 + s] = bp[(size_t)(16 + s) * T_]; }
    if (c0 + 32 < C_)
      __builtin_prefetch(bp + (size_t)32 * T_, 0, 1);   // global_prefetch next K slab

#pragma unroll
    for (int ot = 0; ot < 4; ++ot) {
      // A tile (16xo x 32xc): weights row-major [O,C], K=c contiguous
      const __bf16* ap = W + (size_t)(o0 + ot * 16 + ln16) * C_ + c0 + hi * 8;
      V16U Am;
      Am.h[0] = *(const v8bf*)(ap);
      Am.h[1] = *(const v8bf*)(ap + 16);
      acc[ot] = __builtin_amdgcn_wmma_f32_16x16x32_bf16(
          false, Am.v, false, Bm.v, (short)0, acc[ot], false, false);
    }
  }

#pragma unroll
  for (int ot = 0; ot < 4; ++ot)
#pragma unroll
    for (int r = 0; r < 8; ++r)
      out[((size_t)b * O_ + o0 + ot * 16 + r + hi * 8) * T_ + t0 + ln16] = acc[ot][r];
}

// ---------------------------------------------------------------------------
// Launch
// ---------------------------------------------------------------------------
extern "C" void kernel_launch(void* const* d_in, const int* in_sizes, int n_in,
                              void* d_out, int out_size, void* d_ws, size_t ws_size,
                              hipStream_t stream) {
  const float* x     = (const float*)d_in[0];   // [B,C,T]
  const float* pos   = (const float*)d_in[1];   // [B,C,2]
  const float* heads = (const float*)d_in[2];   // [O,D]
  float* out = (float*)d_out;                   // [B,O,T]

  char* ws = (char*)d_ws;
  size_t off = 0;
  __bf16* xb   = (__bf16*)(ws + off); off += (size_t)B_ * C_ * T_ * 2;  // 64 MB
  __bf16* embb = (__bf16*)(ws + off); off += (size_t)B_ * C_ * D_ * 2;  // 64 MB
  __bf16* ht   = (__bf16*)(ws + off); off += (size_t)D_ * O_ * 2;       // 1 MB
  __bf16* wt   = (__bf16*)(ws + off); off += (size_t)B_ * O_ * C_ * 2;  // 8 MB
  float*  offs = (float*)(ws + off);                                    // 64 KB

  {
    int n4 = (B_ * C_ * T_) / 4;
    k_convert_x<<<(n4 + 255) / 256, 256, 0, stream>>>(x, xb, n4);
  }
  {
    int n = D_ * O_;
    k_heads_t<<<(n + 255) / 256, 256, 0, stream>>>(heads, ht, n);
  }
  k_embed<<<B_ * C_, 256, 0, stream>>>(pos, embb, offs);
  k_scores<<<B_ * (O_ / 16), 256, 0, stream>>>(embb, ht, offs, wt);
  {
    dim3 g(T_ / 128, O_ / 64, B_);
    k_out<<<g, 256, 0, stream>>>(wt, xb, out);
  }
}
